// AttentionHead_90022514524673
// MI455X (gfx1250) — matrix operand order repaired and checked by reading, not compiled, not run on hardware
//
#include <hip/hip_runtime.h>

// ---------------------------------------------------------------------------
// Flash-attention forward: B=16, Lq=Lk=1024, X=Dv=512, fp32 in/out.
// bf16 WMMA (v_wmma_f32_16x16x32_bf16) with online softmax.
// Workgroup = 128 threads (4 wave32). BM=64 q-rows, BN=64 k-rows/iter,
// DSLICE=256 of Dv per workgroup (gridDim.z = 2)  -> 40% less WMMA work
// than DSLICE=128 by halving the QK^T recompute.
// Q A-fragments resident in VGPRs; LDS buffer0 reused Q->K; global_prefetch
// of next K/V tiles; packed bf16 convert.
// ---------------------------------------------------------------------------

typedef __attribute__((ext_vector_type(16))) __bf16 v16bf;
typedef __attribute__((ext_vector_type(8)))  float  v8f;

#define B_      16
#define LQ      1024
#define LK      1024
#define XD      512
#define DV      512

#define BM      64
#define BN      64
#define DSLICE  256
#define NTHREADS 128

// LDS layout (bytes): buffer0 holds Q during init, then K tiles.
#define LDS_B0  0                      // max(BM,BN) x XD bf16 = 65536
#define LDS_VT  65536                  // DSLICE x BN bf16     = 32768
#define LDS_P   98304                  // BM x BN bf16         = 8192
#define LDS_BYTES 106496

union FragBF {
  v16bf v;
  uint4 q[2];
};

__device__ __forceinline__ unsigned int pack2_bf16(float a, float b) {
#if __has_builtin(__builtin_amdgcn_cvt_pk_bf16_f32)
  typedef __attribute__((ext_vector_type(2))) __bf16 v2bf;
  union { v2bf v; unsigned u; } cv;
  cv.v = __builtin_amdgcn_cvt_pk_bf16_f32(a, b);
  return cv.u;
#else
  unsigned ua = __float_as_uint(a), ub = __float_as_uint(b);
  ua += 0x7FFFu + ((ua >> 16) & 1u);   // round-to-nearest-even
  ub += 0x7FFFu + ((ub >> 16) & 1u);
  return (ua >> 16) | (ub & 0xFFFF0000u);
#endif
}

__device__ __forceinline__ unsigned short f2bf(float a) {
  unsigned u = __float_as_uint(a);
  u += 0x7FFFu + ((u >> 16) & 1u);
  return (unsigned short)(u >> 16);
}

extern "C" __global__ __launch_bounds__(NTHREADS)
void fa_fwd(const float* __restrict__ Q, const float* __restrict__ K,
            const float* __restrict__ V, float* __restrict__ O) {
  extern __shared__ char smem[];

  const int tid  = threadIdx.x;
  const int wave = tid >> 5;
  const int lane = tid & 31;
  const int half = lane >> 4;    // 0: lanes 0-15, 1: lanes 16-31
  const int nlo  = lane & 15;

  const int b  = blockIdx.y;
  const int q0 = blockIdx.x * BM;
  const int d0 = blockIdx.z * DSLICE;

  unsigned short* b0_s = (unsigned short*)(smem + LDS_B0);  // Q then K tiles
  unsigned short* vt_s = (unsigned short*)(smem + LDS_VT);
  unsigned short* p_s  = (unsigned short*)(smem + LDS_P);

  // ---- Stage Q tile (BM x XD) fp32 -> bf16 into LDS buffer 0 --------------
  {
    const float4* gq = (const float4*)(Q + ((size_t)b * LQ + q0) * XD);
    uint2* dq = (uint2*)b0_s;
#pragma unroll
    for (int i = 0; i < (BM * XD / 4) / NTHREADS; ++i) {
      int e = i * NTHREADS + tid;
      float4 x = gq[e];
      dq[e] = make_uint2(pack2_bf16(x.x, x.y), pack2_bf16(x.z, x.w));
    }
  }
  __syncthreads();

  // ---- Load all Q A-fragments into registers (resident across K loop) ----
  const int qrow = wave * 16 + nlo;   // A-fragment row owned by this lane
  FragBF qa[XD / 32];
#pragma unroll
  for (int xc = 0; xc < XD / 32; ++xc) {
    const uint4* ap = (const uint4*)(b0_s + (size_t)qrow * XD + xc * 32 + half * 8);
    qa[xc].q[0] = ap[0];
    qa[xc].q[1] = ap[2];              // +32 bytes
  }
  __syncthreads();                    // buffer 0 now free for K tiles

  // ---- Per-wave accumulators ---------------------------------------------
  v8f o_acc[DSLICE / 16];
#pragma unroll
  for (int dt = 0; dt < DSLICE / 16; ++dt)
#pragma unroll
    for (int j = 0; j < 8; ++j) o_acc[dt][j] = 0.0f;

  float m_i[8], l_i[8];
#pragma unroll
  for (int r = 0; r < 8; ++r) { m_i[r] = -3.0e38f; l_i[r] = 0.0f; }

  // ---- Stream over K tiles ------------------------------------------------
  for (int kt = 0; kt < LK / BN; ++kt) {
    const int kbase = kt * BN;

    // Stage K tile (BN x XD) fp32 -> bf16 into buffer 0
    {
      const float4* gk = (const float4*)(K + ((size_t)b * LK + kbase) * XD);
      uint2* dk = (uint2*)b0_s;
#pragma unroll
      for (int i = 0; i < (BN * XD / 4) / NTHREADS; ++i) {
        int e = i * NTHREADS + tid;
        float4 x = gk[e];
        dk[e] = make_uint2(pack2_bf16(x.x, x.y), pack2_bf16(x.z, x.w));
      }
    }
    // Stage V^T tile (DSLICE x BN): transpose during fp32 -> bf16 convert
    {
#pragma unroll
      for (int i = 0; i < (BN * DSLICE / 4) / NTHREADS; ++i) {
        int e4 = i * NTHREADS + tid;
        int kk = e4 / (DSLICE / 4);
        int dl = (e4 % (DSLICE / 4)) * 4;
        float4 x = *(const float4*)(V + ((size_t)b * LK + kbase + kk) * DV + d0 + dl);
        vt_s[(dl + 0) * BN + kk] = f2bf(x.x);
        vt_s[(dl + 1) * BN + kk] = f2bf(x.y);
        vt_s[(dl + 2) * BN + kk] = f2bf(x.z);
        vt_s[(dl + 3) * BN + kk] = f2bf(x.w);
      }
    }

    // Prefetch next iteration's K/V tiles into cache (global_prefetch_b8)
    if (kt + 1 < LK / BN) {
      const float* nk = K + ((size_t)b * LK + kbase + BN) * XD;
#pragma unroll
      for (int i = 0; i < 8; ++i)      // 128KB tile, 128B lines, 128 threads
        __builtin_prefetch(nk + ((size_t)(i * NTHREADS + tid)) * 32, 0, 3);
      const float* nv = V + ((size_t)b * LK + kbase + BN) * DV + d0;
#pragma unroll
      for (int i = 0; i < 4; ++i) {    // 64 rows x 1KB slice = 512 lines
        int line = i * NTHREADS + tid;
        __builtin_prefetch(nv + (size_t)(line >> 3) * DV + (line & 7) * 32, 0, 3);
      }
    }
    __syncthreads();

    // ---- S = Q * K^T : 4 C-tiles (16x16) per wave, K-dim = XD in 32-chunks
    v8f s[BN / 16];
#pragma unroll
    for (int t = 0; t < BN / 16; ++t)
#pragma unroll
      for (int j = 0; j < 8; ++j) s[t][j] = 0.0f;

    for (int xc = 0; xc < XD / 32; ++xc) {
      FragBF bt[BN / 16];   // load all B-frags first, then WMMA group
#pragma unroll
      for (int t = 0; t < BN / 16; ++t) {
        const uint4* bp = (const uint4*)(b0_s + (size_t)(t * 16 + nlo) * XD + xc * 32 + half * 8);
        bt[t].q[0] = bp[0];
        bt[t].q[1] = bp[2];
      }
#pragma unroll
      for (int t = 0; t < BN / 16; ++t)
        s[t] = __builtin_amdgcn_wmma_f32_16x16x32_bf16(
            false, qa[xc].v, false, bt[t].v, (short)0, s[t], false, false);
    }

    // ---- Online softmax update (TEMP = 1.0) -------------------------------
    float alpha[8];
#pragma unroll
    for (int r = 0; r < 8; ++r) {
      float mx = s[0][r];
#pragma unroll
      for (int t = 1; t < BN / 16; ++t) mx = fmaxf(mx, s[t][r]);
      mx = fmaxf(mx, __shfl_xor(mx, 1, 16));
      mx = fmaxf(mx, __shfl_xor(mx, 2, 16));
      mx = fmaxf(mx, __shfl_xor(mx, 4, 16));
      mx = fmaxf(mx, __shfl_xor(mx, 8, 16));
      float mn = fmaxf(m_i[r], mx);
      float al = __expf(m_i[r] - mn);
      float rs = 0.0f;
#pragma unroll
      for (int t = 0; t < BN / 16; ++t) {
        float p = __expf(s[t][r] - mn);
        s[t][r] = p;
        rs += p;
      }
      rs += __shfl_xor(rs, 1, 16);
      rs += __shfl_xor(rs, 2, 16);
      rs += __shfl_xor(rs, 4, 16);
      rs += __shfl_xor(rs, 8, 16);
      l_i[r] = l_i[r] * al + rs;
      m_i[r] = mn;
      alpha[r] = al;
    }

#pragma unroll
    for (int dt = 0; dt < DSLICE / 16; ++dt)
#pragma unroll
      for (int r = 0; r < 8; ++r) o_acc[dt][r] *= alpha[r];

    // ---- Spill P (C-layout f32) to LDS as row-major bf16 ------------------
#pragma unroll
    for (int t = 0; t < BN / 16; ++t)
#pragma unroll
      for (int r = 0; r < 8; ++r) {
        int row = wave * 16 + r + half * 8;
        p_s[(size_t)row * BN + t * 16 + nlo] = f2bf(s[t][r]);
      }
    asm volatile("s_wait_dscnt 0x0" ::: "memory");

    // ---- O += P * V : A = P (16xBN), B = V^T columns, K-dim = BN ----------
    const int prow = wave * 16 + nlo;
#pragma unroll
    for (int c = 0; c < BN / 32; ++c) {
      FragBF pa;
      const uint4* pp = (const uint4*)(p_s + (size_t)prow * BN + c * 32 + half * 8);
      pa.q[0] = pp[0];
      pa.q[1] = pp[2];
#pragma unroll
      for (int dt = 0; dt < DSLICE / 16; ++dt) {
        FragBF vb;
        const uint4* vp = (const uint4*)(vt_s + (size_t)(dt * 16 + nlo) * BN + c * 32 + half * 8);
        vb.q[0] = vp[0];
        vb.q[1] = vp[2];
        o_acc[dt] = __builtin_amdgcn_wmma_f32_16x16x32_bf16(
            false, pa.v, false, vb.v, (short)0, o_acc[dt], false, false);
      }
    }
    __syncthreads();   // before K/V^T restage next iteration
  }

  // ---- Normalize and store ------------------------------------------------
  float rinv[8];
#pragma unroll
  for (int r = 0; r < 8; ++r) rinv[r] = 1.0f / l_i[r];

#pragma unroll
  for (int dt = 0; dt < DSLICE / 16; ++dt)
#pragma unroll
    for (int r = 0; r < 8; ++r) {
      int row  = q0 + wave * 16 + r + half * 8;
      int dcol = d0 + dt * 16 + nlo;
      O[((size_t)b * LQ + row) * DV + dcol] = o_acc[dt][r] * rinv[r];
    }
}

extern "C" void kernel_launch(void* const* d_in, const int* in_sizes, int n_in,
                              void* d_out, int out_size, void* d_ws, size_t ws_size,
                              hipStream_t stream) {
  (void)in_sizes; (void)n_in; (void)out_size; (void)d_ws; (void)ws_size;
  const float* q = (const float*)d_in[0];
  const float* k = (const float*)d_in[1];
  const float* v = (const float*)d_in[2];
  float* o = (float*)d_out;

  hipFuncSetAttribute(reinterpret_cast<const void*>(fa_fwd),
                      hipFuncAttributeMaxDynamicSharedMemorySize, LDS_BYTES);

  dim3 grid(LQ / BM, B_, DV / DSLICE);
  fa_fwd<<<grid, NTHREADS, LDS_BYTES, stream>>>(q, k, v, o);
}
